// CapsuleLayer_34677565948477
// MI455X (gfx1250) — matrix operand order repaired and checked by reading, not compiled, not run on hardware
//
#include <hip/hip_runtime.h>
#include <hip/hip_bf16.h>

typedef __attribute__((ext_vector_type(16))) __bf16 v16bf;
typedef __attribute__((ext_vector_type(8)))  float  v8f;

#define NB   32
#define ICH  128
#define HS   64
#define HO   63
#define NHW  3969        // 63*63
#define OC   1024
#define KK   512         // 128*2*2
#define NCAPS 32
#define JK   32

union ABu { uint4 q[2]; v16bf v; };

// ---------------- pre-pass 1: x [b][ic][h][w] f32 -> xb [b][h][w][ic] bf16 ----------------
__global__ __launch_bounds__(256) void xpose_kernel(const float* __restrict__ x,
                                                    __hip_bfloat16* __restrict__ xb) {
    int bh = blockIdx.x;            // 32*64 blocks
    int b = bh >> 6, h = bh & 63;
    __shared__ __hip_bfloat16 t[ICH][HS + 1];
    const float* src = x + (((size_t)b * ICH) * HS + h) * HS;   // + ic*HS*HS + w
    for (int e = threadIdx.x; e < ICH * HS; e += 256) {
        int ic = e >> 6, w = e & 63;
        t[ic][w] = __float2bfloat16(src[(size_t)ic * (HS * HS) + w]);
    }
    __syncthreads();
    __hip_bfloat16* dst = xb + (((size_t)b * HS + h) * HS) * ICH;  // + w*ICH + ic
    for (int e = threadIdx.x; e < ICH * HS; e += 256) {
        int w = e >> 7, ic = e & 127;
        dst[(size_t)w * ICH + ic] = t[ic][w];
    }
}

// ---------------- pre-pass 2: W [oc][ic][kh][kw] f32 -> Wt [oc][kpos*128+ic] bf16 ----------------
__global__ __launch_bounds__(256) void wconv_kernel(const float* __restrict__ W,
                                                    __hip_bfloat16* __restrict__ Wt) {
    int idx = blockIdx.x * 256 + threadIdx.x;
    if (idx >= OC * KK) return;
    int oc = idx >> 9, k = idx & 511;
    int kpos = k >> 7, ic = k & 127;
    Wt[idx] = __float2bfloat16(W[((size_t)(oc * ICH + ic)) * 4 + kpos]);
}

// ---------------- conv: implicit GEMM, bf16 WMMA, f32 accumulate ----------------
// wave -> 32(oc) x 32(n) tile; grid = 32 b * 32 ocT * 125 nT waves = 16000 blocks * 8 waves
__global__ __launch_bounds__(256) void conv_wmma_kernel(const __hip_bfloat16* __restrict__ xb,
                                                        const __hip_bfloat16* __restrict__ Wt,
                                                        const float* __restrict__ bias,
                                                        __hip_bfloat16* __restrict__ U) {
    const int wid  = blockIdx.x * 8 + (threadIdx.x >> 5);
    const int lane = threadIdx.x & 31;
    int b  = wid / 4000;
    int r  = wid - b * 4000;
    int ot = r / 125;
    int nt = r - ot * 125;
    const int oc0 = ot * 32, n0 = nt * 32;
    const bool lo = lane < 16;
    const int  lm = lane & 15;

    v8f acc[2][2];
#pragma unroll
    for (int a = 0; a < 2; a++)
#pragma unroll
        for (int c = 0; c < 2; c++) acc[a][c] = (v8f)(0.0f);

    // A: lane<16 holds K 0..15, lane>=16 holds K 16..31 (contiguous, matches B)
    const __hip_bfloat16* a0 = Wt + (size_t)(oc0 + lm) * KK + (lo ? 0 : 16);
    const __hip_bfloat16* a1 = a0 + (size_t)16 * KK;

    // B: N = lane%16, 16 contiguous K per lane (lo: K0..15, hi: K16..31)
    size_t rowb[2];
#pragma unroll
    for (int ns = 0; ns < 2; ns++) {
        int n = n0 + ns * 16 + lm;
        if (n > NHW - 1) n = NHW - 1;             // clamp (stores masked later)
        int h = n / HO, w = n - h * HO;
        rowb[ns] = (((size_t)b * HS + h) * HS + w) * ICH + (lo ? 0 : 16);
    }

#pragma unroll
    for (int ks = 0; ks < 16; ks++) {             // K = 16 * 32
        const int kpos = ks >> 2;                 // (kh,kw) block of 128 ic
        const int kh = kpos >> 1, kw = kpos & 1;
        const int icb = (ks & 3) * 32;
        ABu a[2];
#pragma unroll
        for (int os = 0; os < 2; os++) {
            const __hip_bfloat16* p = (os ? a1 : a0) + ks * 32;
            a[os].q[0] = *(const uint4*)(p);
            a[os].q[1] = *(const uint4*)(p + 8);
        }
        const size_t koff = (size_t)(kh * HS + kw) * ICH + icb;
#pragma unroll
        for (int ns = 0; ns < 2; ns++) {
            v16bf bv = *(const v16bf*)(xb + rowb[ns] + koff);
#pragma unroll
            for (int os = 0; os < 2; os++)
                acc[os][ns] = __builtin_amdgcn_wmma_f32_16x16x32_bf16(
                    false, a[os].v, false, bv, (short)0, acc[os][ns], false, false);
        }
    }

    // D layout: N = lane%16, M = vgpr + (lane<16 ? 0 : 8). Fuse bias, store bf16 u[b][oc][n].
#pragma unroll
    for (int ns = 0; ns < 2; ns++) {
        int n = n0 + ns * 16 + lm;
        if (n >= NHW) continue;
#pragma unroll
        for (int os = 0; os < 2; os++) {
            int ocb = oc0 + os * 16 + (lo ? 0 : 8);
#pragma unroll
            for (int rr = 0; rr < 8; rr++) {
                int oc = ocb + rr;
                float v = acc[os][ns][rr] + bias[oc];
                U[((size_t)b * OC + oc) * NHW + n] = __float2bfloat16(v);
            }
        }
    }
}

// ---------------- fused routing step: LDS-resident U tile (254 KB / WGP) ----------------
// Per block (b,i):
//   phase 0: stage u[b,i,:,:] (32 x 3969 bf16) into LDS, vectorized 16B
//   phase 1: o_jk = sum_n c[b,i,n] * u[jk,n]   (mode 0: c = 1/32)
//   phase 2: mode 0/1: normalize o block-locally, Blog[b,i,n] = sum_jk o_jk*u[jk,n]
//            mode 2:   squash(o) -> out[b,i,:]
__global__ __launch_bounds__(256) void route_fused_kernel(const __hip_bfloat16* __restrict__ U,
                                                          const float* __restrict__ C,
                                                          float* __restrict__ Blog,
                                                          float* __restrict__ out,
                                                          int mode) {
    extern __shared__ uint4 UtQ[];                       // 32*3969 bf16 = 254016 B
    __hip_bfloat16* Ut = (__hip_bfloat16*)UtQ;
    __shared__ float sred[JK];
    __shared__ float onorm[JK];

    const int b = blockIdx.x >> 5, i = blockIdx.x & 31;
    const __hip_bfloat16* Ub = U + ((size_t)b * OC + i * JK) * NHW;

    if (threadIdx.x < JK) sred[threadIdx.x] = 0.0f;
    {   // stage tile: (32*3969)/8 = 15876 uint4, exact
        const uint4* src = (const uint4*)Ub;
        for (int e = threadIdx.x; e < (JK * NHW) / 8; e += 256) UtQ[e] = src[e];
    }
    __syncthreads();

    const float* Cb = C + ((size_t)(b * NCAPS) + i) * NHW;
    float acc[JK];
#pragma unroll
    for (int j = 0; j < JK; j++) acc[j] = 0.0f;
    for (int n = threadIdx.x; n < NHW; n += 256) {
        float cv = (mode == 0) ? 0.03125f : Cb[n];
#pragma unroll
        for (int j = 0; j < JK; j++)
            acc[j] += cv * __bfloat162float(Ut[j * NHW + n]);
    }
    // wave butterfly: every lane ends with full-wave sums for all j
#pragma unroll
    for (int j = 0; j < JK; j++) {
        for (int s = 16; s > 0; s >>= 1) acc[j] += __shfl_xor(acc[j], s, 32);
    }
    atomicAdd(&sred[threadIdx.x & 31], acc[threadIdx.x & 31]);  // lane l contributes j=l, per wave
    __syncthreads();

    if (threadIdx.x < JK) {       // first wave finalizes
        float v  = sred[threadIdx.x];
        float sq = v * v;
        for (int s = 16; s > 0; s >>= 1) sq += __shfl_xor(sq, s, 32);
        if (mode == 2) {
            float scale = sq / (1.0f + sq);
            out[(size_t)blockIdx.x * JK + threadIdx.x] = scale * v / (sqrtf(sq) + 1e-6f);
        } else {
            onorm[threadIdx.x] = v / fmaxf(sqrtf(sq), 1e-12f);
        }
    }
    if (mode == 2) return;
    __syncthreads();

    float ol[JK];
#pragma unroll
    for (int j = 0; j < JK; j++) ol[j] = onorm[j];
    float* Bp = Blog + (size_t)blockIdx.x * NHW;     // in-place over C row is safe (block-local)
    for (int n = threadIdx.x; n < NHW; n += 256) {
        float s = 0.0f;
#pragma unroll
        for (int j = 0; j < JK; j++)
            s += ol[j] * __bfloat162float(Ut[j * NHW + n]);
        Bp[n] = s;
    }
}

// ---------------- softmax over i (axis 1), in place: Blog -> C ----------------
__global__ __launch_bounds__(256) void route_softmax_kernel(float* __restrict__ Blog) {
    int idx = blockIdx.x * 256 + threadIdx.x;   // over B*NHW
    if (idx >= NB * NHW) return;
    int b = idx / NHW, n = idx - b * NHW;
    float* p = Blog + (size_t)b * NCAPS * NHW + n;
    float v[NCAPS];
    float mx = -1e30f;
#pragma unroll
    for (int i = 0; i < NCAPS; i++) { v[i] = p[(size_t)i * NHW]; mx = fmaxf(mx, v[i]); }
    float sum = 0.0f;
#pragma unroll
    for (int i = 0; i < NCAPS; i++) { v[i] = __expf(v[i] - mx); sum += v[i]; }
    float inv = 1.0f / sum;
#pragma unroll
    for (int i = 0; i < NCAPS; i++) p[(size_t)i * NHW] = v[i] * inv;
}

extern "C" void kernel_launch(void* const* d_in, const int* in_sizes, int n_in,
                              void* d_out, int out_size, void* d_ws, size_t ws_size,
                              hipStream_t stream) {
    const float* x    = (const float*)d_in[0];
    const float* W    = (const float*)d_in[1];
    const float* bias = (const float*)d_in[2];
    float* out = (float*)d_out;

    char* ws = (char*)d_ws;
    const size_t offXB = 0;                              // 32*64*64*128*2  = 33,554,432
    const size_t offWT = offXB + (size_t)33554432;       // 1024*512*2      =  1,048,576
    const size_t offU  = offWT + (size_t)1048576;        // 32*1024*3969*2  = 260,112,384
    const size_t offBL = offU  + (size_t)260046848;      // 32*32*3969*4    =  16,257,024

    __hip_bfloat16* xb   = (__hip_bfloat16*)(ws + offXB);
    __hip_bfloat16* Wt   = (__hip_bfloat16*)(ws + offWT);
    __hip_bfloat16* U    = (__hip_bfloat16*)(ws + offU);
    float*          Blog = (float*)(ws + offBL);

    const int dynLds = JK * NHW * 2;   // 254016 B < 320 KB LDS per workgroup
    (void)hipFuncSetAttribute(reinterpret_cast<const void*>(route_fused_kernel),
                              hipFuncAttributeMaxDynamicSharedMemorySize, dynLds);

    // 1) layout conversions
    xpose_kernel<<<NB * HS, 256, 0, stream>>>(x, xb);
    wconv_kernel<<<(OC * KK) / 256, 256, 0, stream>>>(W, Wt);

    // 2) conv as implicit GEMM with bf16 WMMA (u = conv + bias, bf16, [b][oc][n])
    conv_wmma_kernel<<<16000, 256, 0, stream>>>(xb, Wt, bias, U);

    const int softmax_blocks = (NB * NHW + 255) / 256;

    // 3) routing iter 1 (b=0 => c uniform): o + normalize + logits, fused in LDS
    route_fused_kernel<<<NB * NCAPS, 256, dynLds, stream>>>(U, Blog, Blog, out, 0);
    route_softmax_kernel<<<softmax_blocks, 256, 0, stream>>>(Blog);

    // 4) routing iter 2
    route_fused_kernel<<<NB * NCAPS, 256, dynLds, stream>>>(U, Blog, Blog, out, 1);
    route_softmax_kernel<<<softmax_blocks, 256, 0, stream>>>(Blog);

    // 5) routing iter 3 + squash -> d_out [B][32][32]
    route_fused_kernel<<<NB * NCAPS, 256, dynLds, stream>>>(U, Blog, Blog, out, 2);
}
